// HyperFlowBlock_12163347382950
// MI455X (gfx1250) — matrix operand order repaired and checked by reading, not compiled, not run on hardware
//
#include <hip/hip_runtime.h>
#include <hip/hip_bf16.h>

// ---------------- types ----------------
typedef __bf16 bf16;
typedef __attribute__((ext_vector_type(16))) __bf16 bfx16;
typedef __attribute__((ext_vector_type(8)))  __bf16 bfx8;
typedef __attribute__((ext_vector_type(8)))  float  fx8;

// ---------------- problem constants ----------------
constexpr int Bn = 4;
constexpr int Nn = 65536;
constexpr int Cc = 256;
constexpr int Ll = 64;
constexpr int Ss = 64;
constexpr int NT = Nn / 64;          // 1024 64-token tiles per batch (kernel 3)
constexpr int TPB = 512;             // tokens per block in kernel 1
constexpr int SUBT = TPB / 64;       // 8 sub-tiles per block
constexpr int NTB = Nn / TPB;        // 128 blocks per batch (kernel 1)

// ---------------- workspace layout (bytes, 256-aligned) ----------------
constexpr size_t OFF_ACC  = 0;                      // B*S*C f32   = 262144
constexpr size_t OFF_WSUM = 262144;                 // B*S f32     = 1024
constexpr size_t OFF_WT1  = 263168;                 // 64x256 bf16 (sw_shared^T)
constexpr size_t OFF_WT2  = OFF_WT1 + 32768;        // 256x64 bf16 (sw_feat^T)
constexpr size_t OFF_WT3  = OFF_WT2 + 32768;        // 64x128 bf16 (sw_asn^T)
constexpr size_t OFF_WT4  = OFF_WT3 + 16384;        // 128x256 bf16 (cm_in^T)
constexpr size_t OFF_WT5  = OFF_WT4 + 65536;        // 256x64 bf16 (cm_out^T)
constexpr size_t OFF_STT  = OFF_WT5 + 32768;        // B*256*64 bf16 (st^T)
constexpr size_t OFF_WBUF = OFF_STT + 131072;       // B*N*64 bf16 (assignment w)

// ---------------- device helpers ----------------
__device__ __forceinline__ fx8 wmma_bf16(bfx16 a, bfx16 b, fx8 c) {
  return __builtin_amdgcn_wmma_f32_16x16x32_bf16(false, a, false, b, (short)0, c, false, false);
}

__device__ __forceinline__ bfx16 mk16(bfx8 lo, bfx8 hi) {
  return __builtin_shufflevector(lo, hi, 0,1,2,3,4,5,6,7,8,9,10,11,12,13,14,15);
}

// A fragment (16x32 bf16, row-major source, rows = M): lanes 0-15 -> K[k0..k0+8)+K[k0+16..k0+24),
// lanes 16-31 -> K[k0+8..k0+16)+K[k0+24..k0+32)
__device__ __forceinline__ bfx16 ldA(const bf16* base, int stride, int mt, int k0, int lane) {
  int m = lane & 15, g = lane >> 4;
  const bf16* p = base + (size_t)(mt * 16 + m) * stride + k0 + g * 8;
  bfx8 lo = *(const bfx8*)p;
  bfx8 hi = *(const bfx8*)(p + 16);
  return mk16(lo, hi);
}

// B fragment (32x16 bf16) from N-major (transposed) source: lane n gets K[k0+8g..+8)+K[k0+16+8g..+8) of column n
__device__ __forceinline__ bfx16 ldB(const bf16* base, int stride, int nt, int k0, int lane) {
  int n = lane & 15, g = lane >> 4;
  const bf16* p = base + (size_t)(nt * 16 + n) * stride + k0 + g * 8;
  bfx8 lo = *(const bfx8*)p;
  bfx8 hi = *(const bfx8*)(p + 16);
  return mk16(lo, hi);
}

__device__ __forceinline__ fx8 splat8(float v) {
  fx8 c;
#pragma unroll
  for (int r = 0; r < 8; ++r) c[r] = v;
  return c;
}

__device__ __forceinline__ fx8 ldC(const float* base, int stride, int mt, int ct, int lane, float addb) {
  int n = lane & 15, g = lane >> 4;
  const float* p = base + (size_t)(mt * 16 + g * 8) * stride + ct * 16 + n;
  fx8 c;
#pragma unroll
  for (int r = 0; r < 8; ++r) c[r] = p[(size_t)r * stride] + addb;
  return c;
}

__device__ __forceinline__ void stD_f32(float* base, int stride, int mt, int ct, int lane, fx8 d) {
  int n = lane & 15, g = lane >> 4;
  float* p = base + (size_t)(mt * 16 + g * 8) * stride + ct * 16 + n;
#pragma unroll
  for (int r = 0; r < 8; ++r) p[(size_t)r * stride] = d[r];
}

// accumulate D into an LDS f32 tile (tile ownership is wave-exclusive -> plain RMW)
__device__ __forceinline__ void accD_f32(float* base, int stride, int mt, int ct, int lane, fx8 d) {
  int n = lane & 15, g = lane >> 4;
  float* p = base + (size_t)(mt * 16 + g * 8) * stride + ct * 16 + n;
#pragma unroll
  for (int r = 0; r < 8; ++r) p[(size_t)r * stride] += d[r];
}

__device__ __forceinline__ void stD_bf16(bf16* base, int stride, int mt, int ct, int lane, fx8 d) {
  int n = lane & 15, g = lane >> 4;
  bf16* p = base + (size_t)(mt * 16 + g * 8) * stride + ct * 16 + n;
#pragma unroll
  for (int r = 0; r < 8; ++r) p[(size_t)r * stride] = (bf16)d[r];
}

// transposed store: per-lane contiguous 8 bf16 -> single 16B ds_store
__device__ __forceinline__ void stD_T_bf16(bf16* base, int stride, int mt, int ct, int lane, fx8 d) {
  int n = lane & 15, g = lane >> 4;
  bf16* p = base + (size_t)(ct * 16 + n) * stride + mt * 16 + g * 8;
  bfx8 v;
#pragma unroll
  for (int r = 0; r < 8; ++r) v[r] = (bf16)d[r];
  *(bfx8*)p = v;
}

__device__ __forceinline__ float quad_sum(float v) {
  v += __shfl_xor(v, 1); v += __shfl_xor(v, 2); return v;
}
__device__ __forceinline__ float quad_max(float v) {
  v = fmaxf(v, __shfl_xor(v, 1)); v = fmaxf(v, __shfl_xor(v, 2)); return v;
}
__device__ __forceinline__ float gelu_f(float x) {
  return 0.5f * x * (1.0f + erff(x * 0.70710678118654752f));
}
__device__ __forceinline__ float sigm_f(float x) { return 1.0f / (1.0f + __expf(-x)); }

// =====================================================================
// Kernel 0: zero accumulators, convert+transpose weights to bf16 N-major
// =====================================================================
__global__ void __launch_bounds__(256) k0_prep(
    const float* __restrict__ swW, const float* __restrict__ sfW,
    const float* __restrict__ saW, const float* __restrict__ ciW,
    const float* __restrict__ coW,
    float* __restrict__ acc, float* __restrict__ wsum,
    bf16* __restrict__ wt1, bf16* __restrict__ wt2, bf16* __restrict__ wt3,
    bf16* __restrict__ wt4, bf16* __restrict__ wt5)
{
  int t = blockIdx.x * blockDim.x + threadIdx.x;
  int nth = gridDim.x * blockDim.x;
  for (int i = t; i < Bn * Ss * Cc; i += nth) acc[i] = 0.f;
  for (int i = t; i < Bn * Ss; i += nth) wsum[i] = 0.f;
  for (int i = t; i < Ll * Cc; i += nth) { int n = i / Cc, k = i % Cc; wt1[n * Cc + k] = (bf16)swW[k * Ll + n]; }
  for (int i = t; i < Cc * Ll; i += nth) { int n = i / Ll, k = i % Ll; wt2[n * Ll + k] = (bf16)sfW[k * Cc + n]; }
  for (int i = t; i < Ss * 128; i += nth) { int n = i / 128, k = i % 128; wt3[n * 128 + k] = (bf16)saW[k * Ss + n]; }
  for (int i = t; i < 128 * Cc; i += nth) { int n = i / Cc, k = i % Cc; wt4[n * Cc + k] = (bf16)ciW[k * 128 + n]; }
  for (int i = t; i < Cc * Ll; i += nth) { int n = i / Ll, k = i % Ll; wt5[n * Ll + k] = (bf16)coW[k * Cc + n]; }
}

// =====================================================================
// Kernel 1: persistent over 512 tokens/block. Per 64-token sub-tile:
//   LN1, shared/feat/asn GEMMs, softmax w, w^T@feat accumulated in LDS.
//   One global-atomic flush per block (8x fewer atomics than per-tile).
// =====================================================================
__global__ void __launch_bounds__(256) k1_tokens(
    const float* __restrict__ x, const float* __restrict__ geo,
    const float* __restrict__ ln1g, const float* __restrict__ ln1b,
    const bf16* __restrict__ wt1, const float* __restrict__ swb,
    const bf16* __restrict__ wt2, const float* __restrict__ sfb,
    const bf16* __restrict__ wt3, const float* __restrict__ sab,
    float* __restrict__ acc, float* __restrict__ wsum, bf16* __restrict__ wbuf)
{
  __shared__ bf16  sh_h[64][264];    // LN1(x) bf16
  __shared__ bf16  sh_cat[64][136];  // [shared | geom] bf16
  __shared__ bf16  sh_ft[256][72];   // feat^T bf16 (c-major)
  __shared__ float sh_lg[64][72];    // logits f32
  __shared__ bf16  sh_wt[64][72];    // w^T bf16 (s-major)
  __shared__ float sh_acc[64][256];  // slice-token partial (block-local)
  __shared__ float sh_ws[64];        // w column sums (block-local)

  const int tid = threadIdx.x, wv = tid >> 5, lane = tid & 31;
  const int b = blockIdx.x >> 7;               // NTB = 128
  const int nbase = (blockIdx.x & (NTB - 1)) * TPB;

  // hint the weight tables into cache (small, shared by all blocks)
  __builtin_prefetch(wt1, 0, 2);
  __builtin_prefetch(wt2, 0, 2);
  __builtin_prefetch(wt3, 0, 2);

  for (int i = tid; i < 64 * 256; i += 256) (&sh_acc[0][0])[i] = 0.f;
  if (tid < 64) sh_ws[tid] = 0.f;
  __syncthreads();

  for (int it = 0; it < SUBT; ++it) {
    const int n0 = nbase + it * 64;

    { // LN1 + geometry staging (4 lanes per row)
      const int row = tid >> 2, q = tid & 3;
      const float* xp = x + ((size_t)(b * Nn + n0 + row)) * Cc + q * 64;
      float v[64]; float s = 0.f, s2 = 0.f;
#pragma unroll
      for (int j = 0; j < 16; ++j) {
        float4 t4 = ((const float4*)xp)[j];
        v[4 * j] = t4.x; v[4 * j + 1] = t4.y; v[4 * j + 2] = t4.z; v[4 * j + 3] = t4.w;
        s  += t4.x + t4.y + t4.z + t4.w;
        s2 += t4.x * t4.x + t4.y * t4.y + t4.z * t4.z + t4.w * t4.w;
      }
      s = quad_sum(s); s2 = quad_sum(s2);
      float m = s * (1.f / 256.f);
      float r = rsqrtf(s2 * (1.f / 256.f) - m * m + 1e-5f);
#pragma unroll
      for (int j = 0; j < 64; ++j) {
        int c = q * 64 + j;
        sh_h[row][c] = (bf16)((v[j] - m) * r * ln1g[c] + ln1b[c]);
      }
      const float* gp = geo + ((size_t)(b * Nn + n0 + row)) * Ll + q * 16;
#pragma unroll
      for (int j = 0; j < 16; ++j) sh_cat[row][64 + q * 16 + j] = (bf16)gp[j];
    }
    __syncthreads();

    { // GEMM1: shared = h @ sw_shared_W (K=256) -> sh_cat[:, :64]
      int mt = wv >> 1, nt0 = (wv & 1) * 2;
#pragma unroll
      for (int jj = 0; jj < 2; ++jj) {
        int nt = nt0 + jj;
        fx8 c = splat8(swb[nt * 16 + (lane & 15)]);
#pragma unroll
        for (int kt = 0; kt < 8; ++kt)
          c = wmma_bf16(ldA(&sh_h[0][0], 264, mt, kt * 32, lane),
                        ldB(wt1, 256, nt, kt * 32, lane), c);
        stD_bf16(&sh_cat[0][0], 136, mt, nt, lane, c);
      }
    }
    __syncthreads();

    { // GEMM2: feat = shared @ sw_feat_W (K=64) -> sh_ft (transposed)
      int mt = wv >> 1, ct0 = (wv & 1) * 8;
      bfx16 a0 = ldA(&sh_cat[0][0], 136, mt, 0, lane);
      bfx16 a1 = ldA(&sh_cat[0][0], 136, mt, 32, lane);
#pragma unroll
      for (int j = 0; j < 8; ++j) {
        int ct = ct0 + j;
        fx8 c = splat8(sfb[ct * 16 + (lane & 15)]);
        c = wmma_bf16(a0, ldB(wt2, 64, ct, 0, lane), c);
        c = wmma_bf16(a1, ldB(wt2, 64, ct, 32, lane), c);
        stD_T_bf16(&sh_ft[0][0], 72, mt, ct, lane, c);
      }
    }

    { // GEMM3: logits = [shared|geo] @ sw_asn_W (K=128) -> sh_lg
      int mt = wv >> 1, nt0 = (wv & 1) * 2;
#pragma unroll
      for (int jj = 0; jj < 2; ++jj) {
        int nt = nt0 + jj;
        fx8 c = splat8(sab[nt * 16 + (lane & 15)]);
#pragma unroll
        for (int kt = 0; kt < 4; ++kt)
          c = wmma_bf16(ldA(&sh_cat[0][0], 136, mt, kt * 32, lane),
                        ldB(wt3, 128, nt, kt * 32, lane), c);
        stD_f32(&sh_lg[0][0], 72, mt, nt, lane, c);
      }
    }
    __syncthreads();

    { // softmax over S=64 (4 lanes per row)
      int row = tid >> 2, q = tid & 3;
      float e[16]; float mx = -3.4e38f;
#pragma unroll
      for (int j = 0; j < 16; ++j) { e[j] = sh_lg[row][q * 16 + j]; mx = fmaxf(mx, e[j]); }
      mx = quad_max(mx);
      float ss = 0.f;
#pragma unroll
      for (int j = 0; j < 16; ++j) { e[j] = __expf(e[j] - mx); ss += e[j]; }
      ss = quad_sum(ss);
      float inv = 1.f / ss;
      bfx8 w0, w1;
#pragma unroll
      for (int j = 0; j < 16; ++j) {
        bf16 wb = (bf16)(e[j] * inv);
        sh_wt[q * 16 + j][row] = wb;            // transposed for GEMM4 A
        if (j < 8) w0[j] = wb; else w1[j - 8] = wb;
      }
      bf16* wp = wbuf + ((size_t)(b * Nn + n0 + row)) * Ss + q * 16;
      *(bfx8*)wp = w0;                          // persisted for pass 2
      *(bfx8*)(wp + 8) = w1;
    }
    __syncthreads();

    if (tid < 64) { // per-slice w column sums -> block-local
      float ss = 0.f;
#pragma unroll
      for (int n = 0; n < 64; ++n) ss += (float)sh_wt[tid][n];
      sh_ws[tid] += ss;
    }

    { // GEMM4: slice partial += w^T @ feat (K=64) into LDS accumulator
      int mt = wv >> 1, ct0 = (wv & 1) * 8;
      bfx16 a0 = ldA(&sh_wt[0][0], 72, mt, 0, lane);
      bfx16 a1 = ldA(&sh_wt[0][0], 72, mt, 32, lane);
#pragma unroll
      for (int j = 0; j < 8; ++j) {
        int ct = ct0 + j;
        fx8 c = splat8(0.f);
        c = wmma_bf16(a0, ldB(&sh_ft[0][0], 72, ct, 0, lane), c);
        c = wmma_bf16(a1, ldB(&sh_ft[0][0], 72, ct, 32, lane), c);
        accD_f32(&sh_acc[0][0], 256, mt, ct, lane, c);
      }
    }
    __syncthreads();
  }

  // single global flush per block
  for (int i = tid; i < 64 * 256; i += 256) {
    int s = i >> 8, c = i & 255;
    atomicAdd(&acc[((size_t)(b * Ss + s)) * Cc + c], sh_acc[s][c]);
  }
  if (tid < 64) atomicAdd(&wsum[b * Ss + tid], sh_ws[tid]);
}

// =====================================================================
// Kernel 2: latent transition on (B,S,L) — tiny, scalar VALU
// =====================================================================
__global__ void __launch_bounds__(256) k2_latent(
    const float* __restrict__ acc, const float* __restrict__ wsum,
    const float* __restrict__ ldW, const float* __restrict__ ldb,
    const float* __restrict__ dng, const float* __restrict__ dnb,
    const float* __restrict__ laW, const float* __restrict__ lab,
    const float* __restrict__ lsW, const float* __restrict__ lcW,
    const float* __restrict__ ong, const float* __restrict__ onb,
    const float* __restrict__ luW, const float* __restrict__ lub,
    bf16* __restrict__ stT)
{
  __shared__ float sk[64][256];
  __shared__ float lsm[64][64];
  __shared__ float awm[64][16];
  __shared__ float anc[16][64];
  __shared__ float ctxm[64][64];
  __shared__ float unm[64][64];
  __shared__ float asum[16];
  const int t = threadIdx.x, b = blockIdx.x;

  for (int i = t; i < 64 * 256; i += 256) {
    int s = i >> 8, c = i & 255;
    float ws = fmaxf(wsum[b * 64 + s], 1e-6f);
    sk[s][c] = acc[((size_t)(b * 64 + s)) * 256 + c] / ws;
  }
  __syncthreads();
  for (int i = t; i < 64 * 64; i += 256) {            // down proj
    int s = i >> 6, d = i & 63;
    float a = ldb[d];
    for (int k = 0; k < 256; ++k) a += sk[s][k] * ldW[k * 64 + d];
    lsm[s][d] = a;
  }
  __syncthreads();
  if (t < 64) {                                       // LN(dn)
    float s1 = 0.f, s2 = 0.f;
    for (int d = 0; d < 64; ++d) { float v = lsm[t][d]; s1 += v; s2 += v * v; }
    float m = s1 / 64.f, r = rsqrtf(s2 / 64.f - m * m + 1e-5f);
    for (int d = 0; d < 64; ++d) lsm[t][d] = (lsm[t][d] - m) * r * dng[d] + dnb[d];
  }
  __syncthreads();
  for (int i = t; i < 64 * 16; i += 256) {            // anchor logits
    int s = i >> 4, a = i & 15;
    float v = lab[a];
    for (int d = 0; d < 64; ++d) v += lsm[s][d] * laW[d * 16 + a];
    awm[s][a] = v;
  }
  __syncthreads();
  if (t < 64) {                                       // softmax over A=16
    float mx = -3.4e38f;
    for (int a = 0; a < 16; ++a) mx = fmaxf(mx, awm[t][a]);
    float ss = 0.f;
    for (int a = 0; a < 16; ++a) { float e = __expf(awm[t][a] - mx); awm[t][a] = e; ss += e; }
    float inv = 1.f / ss;
    for (int a = 0; a < 16; ++a) awm[t][a] *= inv;
  }
  __syncthreads();
  if (t < 16) {
    float ss = 0.f;
    for (int s = 0; s < 64; ++s) ss += awm[s][t];
    asum[t] = fmaxf(ss, 1e-6f);
  }
  __syncthreads();
  for (int i = t; i < 16 * 64; i += 256) {            // anchors
    int a = i >> 6, d = i & 63;
    float v = 0.f;
    for (int s = 0; s < 64; ++s) v += awm[s][a] * lsm[s][d];
    anc[a][d] = v / asum[a];
  }
  __syncthreads();
  for (int i = t; i < 64 * 64; i += 256) {            // ctx
    int s = i >> 6, d = i & 63;
    float v = 0.f;
    for (int a = 0; a < 16; ++a) v += awm[s][a] * anc[a][d];
    ctxm[s][d] = v;
  }
  __syncthreads();
  for (int i = t; i < 64 * 64; i += 256) {            // ul = ls + gelu(ls@Wself + ctx@Wctx)
    int s = i >> 6, d = i & 63;
    float v1 = 0.f, v2 = 0.f;
    for (int k = 0; k < 64; ++k) { v1 += lsm[s][k] * lsW[k * 64 + d]; v2 += ctxm[s][k] * lcW[k * 64 + d]; }
    unm[s][d] = lsm[s][d] + gelu_f(v1 + v2);
  }
  __syncthreads();
  if (t < 64) {                                       // LN(out)
    float s1 = 0.f, s2 = 0.f;
    for (int d = 0; d < 64; ++d) { float v = unm[t][d]; s1 += v; s2 += v * v; }
    float m = s1 / 64.f, r = rsqrtf(s2 / 64.f - m * m + 1e-5f);
    for (int d = 0; d < 64; ++d) unm[t][d] = (unm[t][d] - m) * r * ong[d] + onb[d];
  }
  __syncthreads();
  for (int i = t; i < 64 * 256; i += 256) {           // up proj -> st^T bf16
    int s = i >> 8, c = i & 255;
    float v = lub[c];
    for (int d = 0; d < 64; ++d) v += unm[s][d] * luW[d * 256 + c];
    stT[(size_t)b * Cc * Ll + c * 64 + s] = (bf16)v;
  }
}

// =====================================================================
// Kernel 3: reader (w @ st, residual fused into WMMA C) + LN2 + channel
//           mixer (cm_in GEMM, GLU gate, cm_out GEMM, residual) -> out
// =====================================================================
__global__ void __launch_bounds__(256) k3_reader_cmix(
    const float* __restrict__ x, const bf16* __restrict__ wbuf,
    const bf16* __restrict__ stT,
    const float* __restrict__ ln2g, const float* __restrict__ ln2b,
    const bf16* __restrict__ wt4, const float* __restrict__ cib,
    const bf16* __restrict__ wt5, const float* __restrict__ cob,
    float* __restrict__ out)
{
  __shared__ float xs[64][264];     // x (+reader residual) f32
  __shared__ bf16  sh_h[64][264];   // LN2(x) bf16
  __shared__ float vg[64][136];     // cm_in output f32
  __shared__ bf16  gB[64][72];      // gated bf16

  const int tid = threadIdx.x, wv = tid >> 5, lane = tid & 31;
  const int b = blockIdx.x >> 10;
  const int n0 = (blockIdx.x & (NT - 1)) << 6;

  __builtin_prefetch(wt4, 0, 2);
  __builtin_prefetch(wt5, 0, 2);

  { // stage x tile
    int row = tid >> 2, q = tid & 3;
    const float* xp = x + ((size_t)(b * Nn + n0 + row)) * Cc + q * 64;
#pragma unroll
    for (int j = 0; j < 16; ++j)
      *(float4*)&xs[row][q * 64 + j * 4] = ((const float4*)xp)[j];
  }
  __syncthreads();

  { // GEMM5: x += w @ st (K=64), C seeded from xs
    int mt = wv >> 1, ct0 = (wv & 1) * 8;
    const bf16* wp = wbuf + ((size_t)(b * Nn + n0)) * Ss;
    const bf16* stb = stT + (size_t)b * Cc * Ll;   // [c][s]
    bfx16 a0 = ldA(wp, 64, mt, 0, lane);
    bfx16 a1 = ldA(wp, 64, mt, 32, lane);
#pragma unroll
    for (int j = 0; j < 8; ++j) {
      int ct = ct0 + j;
      fx8 c = ldC(&xs[0][0], 264, mt, ct, lane, 0.f);
      c = wmma_bf16(a0, ldB(stb, 64, ct, 0, lane), c);
      c = wmma_bf16(a1, ldB(stb, 64, ct, 32, lane), c);
      stD_f32(&xs[0][0], 264, mt, ct, lane, c);
    }
  }
  __syncthreads();

  { // LN2 from xs -> sh_h bf16
    int row = tid >> 2, q = tid & 3;
    float v[64]; float s = 0.f, s2 = 0.f;
#pragma unroll
    for (int j = 0; j < 64; ++j) {
      float t = xs[row][q * 64 + j];
      v[j] = t; s += t; s2 += t * t;
    }
    s = quad_sum(s); s2 = quad_sum(s2);
    float m = s * (1.f / 256.f);
    float r = rsqrtf(s2 * (1.f / 256.f) - m * m + 1e-5f);
#pragma unroll
    for (int j = 0; j < 64; ++j) {
      int c = q * 64 + j;
      sh_h[row][c] = (bf16)((v[j] - m) * r * ln2g[c] + ln2b[c]);
    }
  }
  __syncthreads();

  { // GEMM6: vg = LN2(x) @ cm_in_W (K=256) + b
    int mt = wv >> 1, ct0 = (wv & 1) * 4;
#pragma unroll
    for (int j = 0; j < 4; ++j) {
      int ct = ct0 + j;
      fx8 c = splat8(cib[ct * 16 + (lane & 15)]);
#pragma unroll
      for (int kt = 0; kt < 8; ++kt)
        c = wmma_bf16(ldA(&sh_h[0][0], 264, mt, kt * 32, lane),
                      ldB(wt4, 256, ct, kt * 32, lane), c);
      stD_f32(&vg[0][0], 136, mt, ct, lane, c);
    }
  }
  __syncthreads();

  { // GLU gate: gelu(v) * sigmoid(g)
    int row = tid >> 2, q = tid & 3;
#pragma unroll
    for (int j = 0; j < 16; ++j) {
      int jj = q * 16 + j;
      gB[row][jj] = (bf16)(gelu_f(vg[row][jj]) * sigm_f(vg[row][64 + jj]));
    }
  }
  __syncthreads();

  { // GEMM7: out = x + gated @ cm_out_W + b (C seeded from xs + bias)
    int mt = wv >> 1, ct0 = (wv & 1) * 8;
    bfx16 a0 = ldA(&gB[0][0], 72, mt, 0, lane);
    bfx16 a1 = ldA(&gB[0][0], 72, mt, 32, lane);
    int n = lane & 15, g = lane >> 4;
#pragma unroll
    for (int j = 0; j < 8; ++j) {
      int ct = ct0 + j;
      fx8 c = ldC(&xs[0][0], 264, mt, ct, lane, cob[ct * 16 + n]);
      c = wmma_bf16(a0, ldB(wt5, 64, ct, 0, lane), c);
      c = wmma_bf16(a1, ldB(wt5, 64, ct, 32, lane), c);
      float* op = out + ((size_t)(b * Nn + n0 + mt * 16 + g * 8)) * Cc + ct * 16 + n;
#pragma unroll
      for (int r = 0; r < 8; ++r) op[(size_t)r * Cc] = c[r];
    }
  }
}

// =====================================================================
extern "C" void kernel_launch(void* const* d_in, const int* in_sizes, int n_in,
                              void* d_out, int out_size, void* d_ws, size_t ws_size,
                              hipStream_t stream) {
  (void)in_sizes; (void)n_in; (void)out_size; (void)ws_size;
  const float* node = (const float*)d_in[0];
  const float* geo  = (const float*)d_in[1];
  const float* ln1g = (const float*)d_in[2];
  const float* ln1b = (const float*)d_in[3];
  const float* swW  = (const float*)d_in[4];
  const float* swb  = (const float*)d_in[5];
  const float* sfW  = (const float*)d_in[6];
  const float* sfb  = (const float*)d_in[7];
  const float* saW  = (const float*)d_in[8];
  const float* sab  = (const float*)d_in[9];
  const float* ldW  = (const float*)d_in[10];
  const float* ldb  = (const float*)d_in[11];
  const float* dng  = (const float*)d_in[12];
  const float* dnb  = (const float*)d_in[13];
  const float* laW  = (const float*)d_in[14];
  const float* lab  = (const float*)d_in[15];
  const float* lsW  = (const float*)d_in[16];
  const float* lcW  = (const float*)d_in[17];
  const float* ong  = (const float*)d_in[18];
  const float* onb  = (const float*)d_in[19];
  const float* luW  = (const float*)d_in[20];
  const float* lub  = (const float*)d_in[21];
  const float* ln2g = (const float*)d_in[22];
  const float* ln2b = (const float*)d_in[23];
  const float* ciW  = (const float*)d_in[24];
  const float* cib  = (const float*)d_in[25];
  const float* coW  = (const float*)d_in[26];
  const float* cob  = (const float*)d_in[27];

  char* ws = (char*)d_ws;
  float* acc  = (float*)(ws + OFF_ACC);
  float* wsum = (float*)(ws + OFF_WSUM);
  bf16* wt1   = (bf16*)(ws + OFF_WT1);
  bf16* wt2   = (bf16*)(ws + OFF_WT2);
  bf16* wt3   = (bf16*)(ws + OFF_WT3);
  bf16* wt4   = (bf16*)(ws + OFF_WT4);
  bf16* wt5   = (bf16*)(ws + OFF_WT5);
  bf16* stT   = (bf16*)(ws + OFF_STT);
  bf16* wbuf  = (bf16*)(ws + OFF_WBUF);
  float* outp = (float*)d_out;

  hipLaunchKernelGGL(k0_prep, dim3(256), dim3(256), 0, stream,
                     swW, sfW, saW, ciW, coW, acc, wsum, wt1, wt2, wt3, wt4, wt5);
  hipLaunchKernelGGL(k1_tokens, dim3(Bn * NTB), dim3(256), 0, stream,
                     node, geo, ln1g, ln1b, wt1, swb, wt2, sfb, wt3, sab,
                     acc, wsum, wbuf);
  hipLaunchKernelGGL(k2_latent, dim3(Bn), dim3(256), 0, stream,
                     acc, wsum, ldW, ldb, dng, dnb, laW, lab, lsW, lcW,
                     ong, onb, luW, lub, stT);
  hipLaunchKernelGGL(k3_reader_cmix, dim3(Bn * NT), dim3(256), 0, stream,
                     node, wbuf, stT, ln2g, ln2b, wt4, cib, wt5, cob, outp);
}
